// MultiHeadAttentionQuantum_65481071405767
// MI455X (gfx1250) — matrix operand order repaired and checked by reading, not compiled, not run on hardware
//
#include <hip/hip_runtime.h>
#include <hip/hip_bf16.h>

// ---------------------------------------------------------------------------
// Quantum multi-head attention, MI455X (gfx1250, wave32).
//   Kernel 1: out[b,s,e] = quantum measure (cos + group-of-8 cumprod) -> f16
//   Kernel 2: fused flash attention, Q=K=V=out, WMMA f16->f32, online softmax
// ---------------------------------------------------------------------------

#define BATCH 8
#define SEQ   2048
#define EMB   128
#define HEADS 16
#define DK    8
#define BM    64            // query rows per workgroup (16 per wave, 4 waves)
#define BN    64            // key/value rows per inner tile
#define KTILES (SEQ / BN)   // 32
#define VT_STRIDE 72        // padded stride (halves) for transposed V tile

typedef __attribute__((ext_vector_type(16))) _Float16 v16h;
typedef __attribute__((ext_vector_type(8)))  float    v8f;

union F16x16 { v16h v; uint4 u[2]; };
union H8     { uint4 u; _Float16 h[8]; };

// ---- fragment loaders (layouts per CDNA5 ISA 7.12.2) ----------------------

// 16x32 f16 A tile.  lane<16: row=lane, k = k0+0..7 (elems 0..7), k0+16..23
// (elems 8..15).  lane>=16: same rows, k offsets +8.
__device__ __forceinline__ v16h load_a_f16(const _Float16* base, int rowStride, int lane) {
    const int r  = lane & 15;
    const int hi = lane >> 4;
    const _Float16* p = base + r * rowStride + hi * 8;
    F16x16 f;
    f.u[0] = *(const uint4*)(p);        // k0 + hi*8 + 0..7
    f.u[1] = *(const uint4*)(p + 16);   // k0 + 16 + hi*8 + 0..7
    return f.v;
}

// 32x16 f16 B tile.  lane<16: col=lane, K=0..7 (elems 0..7), K=16..23 (elems
// 8..15); lane>=16: col, K offsets +8.  `base` points at (col0,k0); each matrix row of B is a row of the
// (row-major) K tile / transposed-V tile, so 16 halves are contiguous.
__device__ __forceinline__ v16h load_b_f16(const _Float16* base, int colStride, int lane) {
    const int c  = lane & 15;
    const int hi = lane >> 4;
    const _Float16* p = base + c * colStride + hi * 8;
    F16x16 f;
    f.u[0] = *(const uint4*)(p);        // K = hi*8 + 0..7
    f.u[1] = *(const uint4*)(p + 16);   // K = 16 + hi*8 + 0..7
    return f.v;
}

// ---------------------------------------------------------------------------
// Kernel 1: quantum measurement.  One thread per (b,s,h) group of 8 elements.
// c_i = cos(x_i + theta_i); out[0] = c1*...*c7 ; out[w>=1] = c0*...*cw.
// ---------------------------------------------------------------------------
__global__ __launch_bounds__(256)
void qmeasure_kernel(const float* __restrict__ x,
                     const float* __restrict__ theta,
                     _Float16* __restrict__ outh) {
    const int g = blockIdx.x * 256 + threadIdx.x;     // group id, 0..262143
    const float4* xp = (const float4*)(x + (size_t)g * 8);
    const float4 x0 = xp[0];
    const float4 x1 = xp[1];

    float c[8];
    c[0] = __cosf(x0.x + theta[0]);
    c[1] = __cosf(x0.y + theta[1]);
    c[2] = __cosf(x0.z + theta[2]);
    c[3] = __cosf(x0.w + theta[3]);
    c[4] = __cosf(x1.x + theta[4]);
    c[5] = __cosf(x1.y + theta[5]);
    c[6] = __cosf(x1.z + theta[6]);
    c[7] = __cosf(x1.w + theta[7]);

    const float z0 = c[1]*c[2]*c[3]*c[4]*c[5]*c[6]*c[7];

    H8 o;
    o.h[0] = (_Float16)z0;
    float run = c[0];
#pragma unroll
    for (int w = 1; w < 8; ++w) { run *= c[w]; o.h[w] = (_Float16)run; }

    *(uint4*)(outh + (size_t)g * 8) = o.u;
}

// ---------------------------------------------------------------------------
// Kernel 2: fused flash attention.  grid = (SEQ/BM, BATCH), 128 threads.
// Each wave owns a 16-row query strip; iterates over 32 K/V tiles of 64 rows.
// ---------------------------------------------------------------------------
__global__ __launch_bounds__(128)
void flash_attn_kernel(const _Float16* __restrict__ outh,
                       float* __restrict__ dout) {
    __shared__ alignas(16) _Float16 Ksh[BN * EMB];            // 16 KB row-major K tile
    __shared__ alignas(16) _Float16 VTsh[EMB * VT_STRIDE];    // 18 KB transposed V tile
    __shared__ alignas(16) _Float16 Psh[4][16 * BN];          //  8 KB per-wave P scratch

    const int b     = blockIdx.y;
    const int qtile = blockIdx.x;
    const int tid   = threadIdx.x;
    const int wave  = tid >> 5;
    const int lane  = tid & 31;

    const _Float16* Ob = outh + (size_t)b * SEQ * EMB;
    const int q0 = qtile * BM + wave * 16;                    // global query row base

    // Q fragments (persistent), one per 32-wide k-step over E=128.
    v16h aQ[4];
#pragma unroll
    for (int ks = 0; ks < 4; ++ks)
        aQ[ks] = load_a_f16(Ob + (size_t)q0 * EMB + ks * 32, EMB, lane);

    // Accumulators in WMMA C layout: acc[n][r] = D[r + hi*8][n*16 + lane%16].
    v8f acc[8], m8, l8;
#pragma unroll
    for (int n = 0; n < 8; ++n)
#pragma unroll
        for (int e = 0; e < 8; ++e) acc[n][e] = 0.0f;
#pragma unroll
    for (int e = 0; e < 8; ++e) { m8[e] = -1e30f; l8[e] = 0.0f; }

    const float scale = 0.35355339059327373f;                 // 1/sqrt(8)

    for (int kt = 0; kt < KTILES; ++kt) {
        __syncthreads();                                      // prior tile reads done
        // --- cooperative stage of K (row-major) and V (transposed) tiles ---
        const uint4* gK = (const uint4*)(Ob + (size_t)kt * BN * EMB); // 1024 uint4
        uint4*       sK = (uint4*)Ksh;
#pragma unroll
        for (int i = 0; i < 8; ++i) {
            const int pos = tid + i * 128;
            H8 t; t.u = gK[pos];
            sK[pos] = t.u;
            const int key = pos >> 4;                         // row within tile
            const int e0  = (pos & 15) << 3;                  // embedding col base
#pragma unroll
            for (int j = 0; j < 8; ++j)
                VTsh[(e0 + j) * VT_STRIDE + key] = t.h[j];
        }
        if (kt + 1 < KTILES)                                  // global_prefetch next tile
            __builtin_prefetch((const void*)(Ob + (size_t)(kt + 1) * BN * EMB), 0, 1);
        __syncthreads();                                      // tile ready

        // --- scores: S = Q * K^T * scale  (4 N-subtiles x 4 k-steps) ---
        v8f sc[4];
#pragma unroll
        for (int n = 0; n < 4; ++n) {
            v8f c;
#pragma unroll
            for (int e = 0; e < 8; ++e) c[e] = 0.0f;
#pragma unroll
            for (int ks = 0; ks < 4; ++ks) {
                v16h bK = load_b_f16(&Ksh[(n * 16) * EMB + ks * 32], EMB, lane);
                c = __builtin_amdgcn_wmma_f32_16x16x32_f16(
                        false, aQ[ks], false, bK, (short)0, c, false, false);
            }
#pragma unroll
            for (int e = 0; e < 8; ++e) sc[n][e] = c[e] * scale;
        }

        // --- online softmax: row max over 64 cols (lane-local then butterfly) ---
        v8f tm;
#pragma unroll
        for (int e = 0; e < 8; ++e)
            tm[e] = fmaxf(fmaxf(sc[0][e], sc[1][e]), fmaxf(sc[2][e], sc[3][e]));
#pragma unroll
        for (int off = 1; off < 16; off <<= 1)
#pragma unroll
            for (int e = 0; e < 8; ++e)
                tm[e] = fmaxf(tm[e], __shfl_xor(tm[e], off, 32));

        v8f mnew, alpha;
#pragma unroll
        for (int e = 0; e < 8; ++e) {
            mnew[e]  = fmaxf(m8[e], tm[e]);
            alpha[e] = __expf(m8[e] - mnew[e]);
        }
#pragma unroll
        for (int n = 0; n < 4; ++n)
#pragma unroll
            for (int e = 0; e < 8; ++e)
                sc[n][e] = __expf(sc[n][e] - mnew[e]);

        v8f ts;
#pragma unroll
        for (int e = 0; e < 8; ++e) ts[e] = sc[0][e] + sc[1][e] + sc[2][e] + sc[3][e];
#pragma unroll
        for (int off = 1; off < 16; off <<= 1)
#pragma unroll
            for (int e = 0; e < 8; ++e) ts[e] += __shfl_xor(ts[e], off, 32);

#pragma unroll
        for (int e = 0; e < 8; ++e) { l8[e] = l8[e] * alpha[e] + ts[e]; m8[e] = mnew[e]; }
#pragma unroll
        for (int n = 0; n < 8; ++n)
#pragma unroll
            for (int e = 0; e < 8; ++e) acc[n][e] *= alpha[e];

        // --- C-layout -> A-layout for P via per-wave private LDS scratch ---
        {
            const int hi = lane >> 4, cc = lane & 15;
            _Float16* pw = &Psh[wave][0];
#pragma unroll
            for (int n = 0; n < 4; ++n)
#pragma unroll
                for (int e = 0; e < 8; ++e)
                    pw[(e + hi * 8) * BN + n * 16 + cc] = (_Float16)sc[n][e];
        }   // same-wave LDS ops are in-order: no barrier needed

        // --- PV: acc += P * V   (2 k-steps over 64 keys, 8 output subtiles) ---
#pragma unroll
        for (int ks = 0; ks < 2; ++ks) {
            v16h aP = load_a_f16(&Psh[wave][ks * 32], BN, lane);
#pragma unroll
            for (int n = 0; n < 8; ++n) {
                v16h bV = load_b_f16(&VTsh[(n * 16) * VT_STRIDE + ks * 32], VT_STRIDE, lane);
                acc[n] = __builtin_amdgcn_wmma_f32_16x16x32_f16(
                             false, aP, false, bV, (short)0, acc[n], false, false);
            }
        }
    }

    // --- normalize and store (f32) ---
    {
        const int hi = lane >> 4, cc = lane & 15;
        float* drow = dout + ((size_t)b * SEQ + q0) * EMB;
#pragma unroll
        for (int e = 0; e < 8; ++e) {
            const float inv = 1.0f / l8[e];
            const int row = e + hi * 8;
#pragma unroll
            for (int n = 0; n < 8; ++n)
                drow[(size_t)row * EMB + n * 16 + cc] = acc[n][e] * inv;
        }
    }
}

// ---------------------------------------------------------------------------
extern "C" void kernel_launch(void* const* d_in, const int* in_sizes, int n_in,
                              void* d_out, int out_size, void* d_ws, size_t ws_size,
                              hipStream_t stream) {
    const float* x     = (const float*)d_in[0];   // [8,2048,128] f32
    const float* theta = (const float*)d_in[1];   // [8] f32
    float*       out   = (float*)d_out;           // [8,2048,128] f32
    _Float16*    outh  = (_Float16*)d_ws;         // 4 MB f16 measured tensor

    const int groups = BATCH * SEQ * HEADS;       // 262144
    qmeasure_kernel<<<groups / 256, 256, 0, stream>>>(x, theta, outh);

    dim3 grid(SEQ / BM, BATCH);                   // 32 x 8 = 256 workgroups
    flash_attn_kernel<<<grid, 128, 0, stream>>>(outh, out);
}
